// MHA_8916352106627
// MI455X (gfx1250) — matrix operand order repaired and checked by reading, not compiled, not run on hardware
//
#include <hip/hip_runtime.h>
#include <stdint.h>

// MI455X / gfx1250 MHA (ViT shape: B=64,S=197,E=768,H=12,HD=64), fp32 in/out.
// bf16 WMMA (v_wmma_f32_16x16x32_bf16) for all matmuls; fp32 softmax.
// GEMM tiles staged with GLOBAL_LOAD_ASYNC_TO_LDS_B128 (double-buffered,
// ASYNCcnt-synchronized). V is written pre-transposed per head (d-major ->
// key-major) so attention PV B-fragments are contiguous wide loads.
// Workspace: bf16 x, 4 weights, Q, K, O, Vt = ~104 MB.

#define DI __device__ __forceinline__

typedef __attribute__((ext_vector_type(16))) __bf16 v16bf;
typedef __attribute__((ext_vector_type(8)))  float  v8f;
typedef unsigned short ushort_t;
typedef unsigned int   uint_t;

union FragB16 { v16bf v; uint_t u[8]; };

constexpr int VLD   = 224;          // Vt row stride (keys), padded & even
constexpr int VT_BH = 64 * VLD;     // Vt elements per (b,h)

DI ushort_t f2bf(float f) {
  union { float f; uint_t u; } x; x.f = f;
  uint_t r = x.u + 0x7FFFu + ((x.u >> 16) & 1u);   // round-to-nearest-even
  return (ushort_t)(r >> 16);
}
DI uint_t pack2bf(float a, float b) {
  return (uint_t)f2bf(a) | ((uint_t)f2bf(b) << 16);
}
DI v8f zero8() {
  v8f z;
#pragma unroll
  for (int i = 0; i < 8; ++i) z[i] = 0.f;
  return z;
}
DI v8f wmma_bf16(const FragB16& a, const FragB16& b, v8f c) {
  return __builtin_amdgcn_wmma_f32_16x16x32_bf16(false, a.v, false, b.v,
                                                 (short)0, c, false, false);
}

// CDNA5 async DMA: global -> LDS, 16 bytes per lane, tracked by ASYNCcnt.
DI void async_ld128(uint_t lds_addr, const void* gaddr) {
  asm volatile("global_load_async_to_lds_b128 %0, %1, off"
               :: "v"(lds_addr), "v"(gaddr) : "memory");
}

// ---------------------------------------------------------------------------
// fp32 -> bf16 pre-conversion (8 elements / thread).
// ---------------------------------------------------------------------------
__global__ void __launch_bounds__(256)
cvt_bf16(const float* __restrict__ src, ushort_t* __restrict__ dst, int n8) {
  const int i = blockIdx.x * blockDim.x + threadIdx.x;
  if (i < n8) {
    const float4 f0 = ((const float4*)src)[2 * i];
    const float4 f1 = ((const float4*)src)[2 * i + 1];
    uint4 o;
    o.x = pack2bf(f0.x, f0.y); o.y = pack2bf(f0.z, f0.w);
    o.z = pack2bf(f1.x, f1.y); o.w = pack2bf(f1.z, f1.w);
    ((uint4*)dst)[i] = o;
  }
}

// ---------------------------------------------------------------------------
// GEMM: C[z] = A(12608x768,bf16) @ W[z]^T(768x768,bf16) + b[z]
// Block tile 64(M) x 128(N), BK=32, 256 threads = 8 waves, each wave 32x32.
// Double-buffered LDS, async global->LDS staging.
// z==2 (V) is stored transposed per head: Vt[bh][d][key], stride VLD.
// ---------------------------------------------------------------------------
template <bool OUT_BF16>
__global__ void __launch_bounds__(256)
gemm_wmma(const ushort_t* __restrict__ Ab,
          const ushort_t* __restrict__ W0, const ushort_t* __restrict__ W1,
          const ushort_t* __restrict__ W2,
          const float* __restrict__ b0, const float* __restrict__ b1,
          const float* __restrict__ b2,
          void* __restrict__ C0, void* __restrict__ C1, void* __restrict__ C2)
{
  constexpr int K = 768, N = 768;
  constexpr int LDT = 40;             // padded LDS row stride (80B, 16B-aligned)
  constexpr int NK = K / 32;          // 24 k-steps

  const int z = blockIdx.z;
  const ushort_t* W  = (z == 0) ? W0 : (z == 1) ? W1 : W2;
  const float* bias  = (z == 0) ? b0 : (z == 1) ? b1 : b2;
  void*        Cp    = (z == 0) ? C0 : (z == 1) ? C1 : C2;

  const int bm0 = blockIdx.y * 64;
  const int bn0 = blockIdx.x * 128;
  const int tid  = threadIdx.x;
  const int lane = tid & 31, wid = tid >> 5;
  const int hf   = lane >> 4, r = lane & 15;
  const int waveM = wid & 1, waveN = wid >> 1;

  __shared__ ushort_t As[2][64][LDT];
  __shared__ ushort_t Bs[2][128][LDT];

  v8f acc[2][2];
#pragma unroll
  for (int i = 0; i < 2; ++i)
#pragma unroll
    for (int j = 0; j < 2; ++j) acc[i][j] = zero8();

  // per-stage async issue: 1x A chunk + 2x B chunks per thread (3 per wave)
  auto issue = [&](int ki, int stg) {
    const int kc = ki * 32;
    {
      const int row = tid >> 2, ch = tid & 3;      // A: 64 rows x 4 16B chunks
      const uint_t lds = (uint_t)(uintptr_t)&As[stg][row][ch * 8];
      async_ld128(lds, &Ab[(size_t)(bm0 + row) * K + kc + ch * 8]);
    }
#pragma unroll
    for (int q = 0; q < 2; ++q) {                  // B: 128 rows x 4 chunks
      const int idx = tid + 256 * q;
      const int row = idx >> 2, ch = idx & 3;
      const uint_t lds = (uint_t)(uintptr_t)&Bs[stg][row][ch * 8];
      async_ld128(lds, &W[(size_t)(bn0 + row) * K + kc + ch * 8]);
    }
  };

  issue(0, 0);
  for (int ki = 0; ki < NK; ++ki) {
    const int cur = ki & 1;
    if (ki + 1 < NK) {
      issue(ki + 1, cur ^ 1);
      asm volatile("s_wait_asynccnt 3" ::: "memory");   // stage ki landed
    } else {
      asm volatile("s_wait_asynccnt 0" ::: "memory");
    }
    __syncthreads();

    FragB16 af[2], bf[2];
#pragma unroll
    for (int t = 0; t < 2; ++t) {                  // A frags: lane=row, K=hf*16+2j
      const int mr = 32 * waveM + 16 * t + r;
#pragma unroll
      for (int j = 0; j < 8; ++j) {
        const int koff = hf * 16 + 2 * j;
        af[t].u[j] = *(const uint_t*)&As[cur][mr][koff];
      }
    }
#pragma unroll
    for (int t = 0; t < 2; ++t) {                  // B frags: lane=col, K=hf*16+2j
      const int nr = 32 * waveN + 16 * t + r;
#pragma unroll
      for (int j = 0; j < 8; ++j) {
        const int koff = hf * 16 + 2 * j;
        bf[t].u[j] = *(const uint_t*)&Bs[cur][nr][koff];
      }
    }
#pragma unroll
    for (int tm = 0; tm < 2; ++tm)
#pragma unroll
      for (int tn = 0; tn < 2; ++tn)
        acc[tm][tn] = wmma_bf16(af[tm], bf[tn], acc[tm][tn]);
    __syncthreads();                               // buffer reuse fence
  }

  // C layout: lane holds col n=r (per half), VGPR v holds row m = v + 8*hf
#pragma unroll
  for (int tm = 0; tm < 2; ++tm) {
#pragma unroll
    for (int tn = 0; tn < 2; ++tn) {
      const int m0 = bm0 + 32 * waveM + 16 * tm + 8 * hf;
      const int n  = bn0 + 32 * waveN + 16 * tn + r;
      const float bb = bias[n];
      if constexpr (OUT_BF16) {
        if (z == 2) {
          // transposed per-head V store: (b, srow, n) -> Vt[b*12+h][d][s]
          // using 768 % 64 == 0: d = n & 63, u = srow*12 + (n>>6),
          // h = u / 197, s = u % 197.
          const int d = n & 63, nn6 = n >> 6;
#pragma unroll
          for (int v = 0; v < 8; ++v) {
            const int m = m0 + v;
            const int b = m / 197, srow = m - b * 197;
            const int u = srow * 12 + nn6;
            const int h = u / 197, s = u - h * 197;
            ((ushort_t*)Cp)[(size_t)(b * 12 + h) * VT_BH + (size_t)d * VLD + s]
                = f2bf(acc[tm][tn][v] + bb);
          }
        } else {
#pragma unroll
          for (int v = 0; v < 8; ++v)
            ((ushort_t*)Cp)[(size_t)(m0 + v) * N + n] = f2bf(acc[tm][tn][v] + bb);
        }
      } else {
#pragma unroll
        for (int v = 0; v < 8; ++v)
          ((float*)Cp)[(size_t)(m0 + v) * N + n] = acc[tm][tn][v] + bb;
      }
    }
  }
}

// ---------------------------------------------------------------------------
// Attention: one wave per (b,h, 16-row q tile). grid = (13, B*H), block = 32.
// Q/K/O are bf16 in the flat [B,S,E] buffer; head (b,h) base = (b*H+h)*12608
// with contiguous 64-elem rows (the reference's transpose-free reshape).
// V comes pre-transposed: Vt[bh][d][key], stride VLD.
// energy -> softmax (exact, fp32) -> /sqrt(768) -> @V.
// ---------------------------------------------------------------------------
__global__ void __launch_bounds__(32)
attn_wmma(const ushort_t* __restrict__ Qb, const ushort_t* __restrict__ Kb,
          const ushort_t* __restrict__ Vt, ushort_t* __restrict__ Ob)
{
  constexpr int S = 197;
  constexpr int NKT = 13;                      // ceil(197/16) key tiles
  const int qt = blockIdx.x;
  const size_t base  = (size_t)blockIdx.y * (size_t)(S * 64);
  const size_t vbase = (size_t)blockIdx.y * (size_t)VT_BH;
  const int lane = threadIdx.x & 31;
  const int hf = lane >> 4, r = lane & 15;
  const int qbase = qt * 16;

  __shared__ ushort_t Pl[16][228];             // P tile (bf16), padded stride

  // zero pad columns 208..227 (keys 208..223 used by last K=32 PV chunk)
  {
    const int c0 = 208 + hf * 10;
#pragma unroll
    for (int c = 0; c < 10; ++c) Pl[r][c0 + c] = 0;
  }

  // Q fragments (rows clamped: OOB q-rows compute garbage but are never stored)
  int qr = qbase + r; if (qr > 196) qr = 196;
  FragB16 qf[2];
#pragma unroll
  for (int c = 0; c < 2; ++c)
#pragma unroll
    for (int j = 0; j < 8; ++j) {
      const int koff = hf * 16 + 2 * j;
      qf[c].u[j] = *(const uint_t*)&Qb[base + (size_t)qr * 64 + 32 * c + koff];
    }

  // S = Q @ K^T, 13 tiles of 16 keys kept in registers (13 x v8f)
  v8f sacc[NKT];
#pragma unroll
  for (int kt = 0; kt < NKT; ++kt) {
    int keyr = kt * 16 + r; if (keyr > 196) keyr = 196;
    v8f s = zero8();
#pragma unroll
    for (int c = 0; c < 2; ++c) {
      FragB16 kf;                               // B frag: col=key, K-dim = d
#pragma unroll
      for (int j = 0; j < 8; ++j) {
        const int koff = hf * 16 + 2 * j;
        kf.u[j] = *(const uint_t*)&Kb[base + (size_t)keyr * 64 + 32 * c + koff];
      }
      s = wmma_bf16(qf[c], kf, s);
    }
    if (kt * 16 + r >= 197) {
#pragma unroll
      for (int v = 0; v < 8; ++v) s[v] = -1e30f;
    }
    sacc[kt] = s;
  }

  // row softmax; rows live per (v,hf), key-cols per lane -> 16-lane reductions
#pragma unroll
  for (int v = 0; v < 8; ++v) {
    float mx = -3.4e38f;
#pragma unroll
    for (int kt = 0; kt < NKT; ++kt) mx = fmaxf(mx, sacc[kt][v]);
#pragma unroll
    for (int off = 8; off >= 1; off >>= 1) mx = fmaxf(mx, __shfl_xor(mx, off, 16));
    float sum = 0.f;
#pragma unroll
    for (int kt = 0; kt < NKT; ++kt) {
      const float e = __expf(sacc[kt][v] - mx);
      sacc[kt][v] = e; sum += e;
    }
#pragma unroll
    for (int off = 8; off >= 1; off >>= 1) sum += __shfl_xor(sum, off, 16);
    const float sc = 1.0f / (sum * 27.712812921102035f);  // post-softmax /sqrt(768)
#pragma unroll
    for (int kt = 0; kt < NKT; ++kt) sacc[kt][v] *= sc;
  }

  // spill P (C layout) to LDS so it can be re-gathered in A layout
#pragma unroll
  for (int kt = 0; kt < NKT; ++kt)
#pragma unroll
    for (int v = 0; v < 8; ++v)
      Pl[v + 8 * hf][kt * 16 + r] = f2bf(sacc[kt][v]);
  __syncthreads();

  // O = P @ V, 7 chunks of K=32 keys (tail keys have P==0; Vt pad harmless)
  v8f oacc[4];
#pragma unroll
  for (int t = 0; t < 4; ++t) oacc[t] = zero8();
#pragma unroll
  for (int kc = 0; kc < 7; ++kc) {
    FragB16 pf;
#pragma unroll
    for (int j = 0; j < 8; ++j) {
      const int koff = hf * 16 + 2 * j;
      pf.u[j] = *(const uint_t*)&Pl[r][kc * 32 + koff];
    }
#pragma unroll
    for (int tn = 0; tn < 4; ++tn) {
      FragB16 vf;  // B frag: col n = d (lane), K-dim = key -> contiguous in Vt
      const size_t vrow = vbase + (size_t)(tn * 16 + r) * VLD + kc * 32 + hf * 16;
#pragma unroll
      for (int j = 0; j < 8; ++j)
        vf.u[j] = *(const uint_t*)&Vt[vrow + 2 * j];
      oacc[tn] = wmma_bf16(pf, vf, oacc[tn]);
    }
  }

  // store valid O rows
#pragma unroll
  for (int tn = 0; tn < 4; ++tn)
#pragma unroll
    for (int v = 0; v < 8; ++v) {
      const int srow = qbase + v + 8 * hf;
      if (srow < 197)
        Ob[base + (size_t)srow * 64 + tn * 16 + r] = f2bf(oacc[tn][v]);
    }
}

// ---------------------------------------------------------------------------
extern "C" void kernel_launch(void* const* d_in, const int* in_sizes, int n_in,
                              void* d_out, int out_size, void* d_ws, size_t ws_size,
                              hipStream_t stream) {
  const float* x  = (const float*)d_in[0];
  const float* Wq = (const float*)d_in[1];
  const float* bq = (const float*)d_in[2];
  const float* Wk = (const float*)d_in[3];
  const float* bk = (const float*)d_in[4];
  const float* Wv = (const float*)d_in[5];
  const float* bv = (const float*)d_in[6];
  const float* Wp = (const float*)d_in[7];
  const float* bp = (const float*)d_in[8];

  const size_t MSZ = (size_t)12608 * 768;      // x / Q / K / O elements
  const size_t WSZ = (size_t)768 * 768;        // weight elements
  const size_t VSZ = (size_t)768 * VT_BH;      // transposed V elements

  ushort_t* xb  = (ushort_t*)d_ws;             // bf16 staging in workspace
  ushort_t* Wqb = xb  + MSZ;
  ushort_t* Wkb = Wqb + WSZ;
  ushort_t* Wvb = Wkb + WSZ;
  ushort_t* Wpb = Wvb + WSZ;
  ushort_t* Qb  = Wpb + WSZ;
  ushort_t* Kb  = Qb  + MSZ;
  ushort_t* Ob  = Kb  + MSZ;
  ushort_t* Vtb = Ob  + MSZ;

  // one-pass fp32 -> bf16 conversions
  const int x8 = (int)(MSZ / 8), w8 = (int)(WSZ / 8);
  cvt_bf16<<<(x8 + 255) / 256, 256, 0, stream>>>(x,  xb,  x8);
  cvt_bf16<<<(w8 + 255) / 256, 256, 0, stream>>>(Wq, Wqb, w8);
  cvt_bf16<<<(w8 + 255) / 256, 256, 0, stream>>>(Wk, Wkb, w8);
  cvt_bf16<<<(w8 + 255) / 256, 256, 0, stream>>>(Wv, Wvb, w8);
  cvt_bf16<<<(w8 + 255) / 256, 256, 0, stream>>>(Wp, Wpb, w8);

  // Fused Q/K/V projections (grid.z selects which; V stored transposed).
  gemm_wmma<true><<<dim3(6, 197, 3), 256, 0, stream>>>(
      xb, Wqb, Wkb, Wvb, bq, bk, bv, Qb, Kb, Vtb);

  // Attention per (b,h) pair, 13 q-tiles of 16 rows each.
  attn_wmma<<<dim3(13, 64 * 12), 32, 0, stream>>>(Qb, Kb, Vtb, Ob);

  // Output projection -> fp32 d_out.
  gemm_wmma<false><<<dim3(6, 197, 1), 256, 0, stream>>>(
      Ob, Wpb, Wpb, Wpb, bp, bp, bp, d_out, d_out, d_out);
}
